// MixtureOfExperts_55551107007028
// MI455X (gfx1250) — matrix operand order repaired and checked by reading, not compiled, not run on hardware
//
#include <hip/hip_runtime.h>
#include <hip/hip_bf16.h>

// Problem dims (fixed by the reference)
#define B_    512
#define D_    2048
#define H1_   4096
#define H2_   4096
#define DO_   1024
#define K_    8
#define GH_   256
#define SZ_   128           // DO_/K_
#define CG_ITERS 24
#define POW_ITERS 64

typedef __attribute__((ext_vector_type(16))) __bf16 v16bf;
typedef __attribute__((ext_vector_type(2)))  __bf16 v2bf;
typedef __attribute__((ext_vector_type(8)))  float  v8f;

union Frag32B { uint4 q[2]; v16bf v; };

__device__ __forceinline__ unsigned short f2bf(float f) {
  unsigned u = __builtin_bit_cast(unsigned, f);
  u += 0x7fffu + ((u >> 16) & 1u);     // round-to-nearest-even
  return (unsigned short)(u >> 16);
}
__device__ __forceinline__ unsigned pack2(float a, float b) {
#if __has_builtin(__builtin_amdgcn_cvt_pk_bf16_f32)
  v2bf r = __builtin_amdgcn_cvt_pk_bf16_f32(a, b);
  return __builtin_bit_cast(unsigned, r);
#else
  return (unsigned)f2bf(a) | ((unsigned)f2bf(b) << 16);
#endif
}

// low 32 bits of a generic pointer to __shared__ == LDS byte address (ISA 10.2)
__device__ __forceinline__ unsigned lds_addr(const void* p) {
  return (unsigned)(unsigned long long)p;
}

// ---------------------------------------------------------------------------
// Generic tiled WMMA GEMM: C = act(opA(MxK) * opW(KxN) + bias)
//   aMode: 0 = fp32 row-major (convert in regs), 1 = fp32 transposed,
//          2 = bf16 row-major -> global_load_async_to_lds_b128 (DMA, no VALU)
//   outBF16: store C as bf16 (feeds the next GEMM's aMode==2 path)
// Tile 128x128x32; 256 threads = 8 waves (2x4); wave = 64x32 (4x2 WMMA tiles).
// All call sites have M%128==0, N%128==0, K%32==0.
// ---------------------------------------------------------------------------
#define TM 128
#define TN 128
#define TK 32
#define LS 48   // LDS row stride (bf16 elems): 96B rows, 16B aligned

__global__ __launch_bounds__(256) void wmma_gemm_kernel(
    const void* __restrict__ Ab, long strideA, int lda, int aMode,
    const float* __restrict__ Wb, long strideW, int ldw, int transW,
    const float* __restrict__ biasb, long strideBias,
    void* __restrict__ Cb, long strideC, int ldc,
    int M, int N, int Kdim, int relu, int accum, int outBF16)
{
  __shared__ unsigned short sA[2][TM * LS];
  __shared__ unsigned short sB[2][TN * LS];

  const float* W = Wb + (long)blockIdx.z * strideW;
  const float* bias = biasb ? (biasb + (long)blockIdx.z * strideBias) : nullptr;

  const int m0 = blockIdx.y * TM;
  const int n0 = blockIdx.x * TN;
  const int tid  = threadIdx.x;
  const int lane = tid & 31;
  const int wave = tid >> 5;
  const int wm = (wave >> 2) * 64;   // 0 or 64
  const int wn = (wave & 3) * 32;    // 0,32,64,96

  v8f acc[4][2] = {};

  auto loadA = [&](int buf, int kt) {
    if (aMode == 2) {                           // bf16 row-major: async DMA -> LDS
      const unsigned short* A16 = (const unsigned short*)Ab + (long)blockIdx.z * strideA;
      #pragma unroll
      for (int c = 0; c < 2; ++c) {
        int f8 = tid + c * 256;                 // 512 chunks of 8 bf16 (16B)
        int m  = f8 >> 2;
        int kk = (f8 & 3) * 8;
        const unsigned short* g = &A16[(long)(m0 + m) * lda + kt + kk];
        unsigned dst = lds_addr(&sA[buf][m * LS + kk]);
        asm volatile("global_load_async_to_lds_b128 %0, %1, off"
                     :: "v"(dst), "v"((unsigned long long)g) : "memory");
      }
    } else if (aMode == 0) {                    // fp32 row-major, convert in regs
      const float* A = (const float*)Ab + (long)blockIdx.z * strideA;
      #pragma unroll
      for (int c = 0; c < 4; ++c) {
        int f4 = tid + c * 256;                 // 1024 float4 = 128x32
        int m  = f4 >> 3;
        int kk = (f4 & 7) * 4;
        const float4 v = *(const float4*)&A[(long)(m0 + m) * lda + kt + kk];
        *(uint2*)&sA[buf][m * LS + kk] = make_uint2(pack2(v.x, v.y), pack2(v.z, v.w));
      }
    } else {                                    // fp32, A(m,kk) = Ab[kk*lda + m]
      const float* A = (const float*)Ab + (long)blockIdx.z * strideA;
      #pragma unroll
      for (int c = 0; c < 4; ++c) {
        int f4 = tid + c * 256;
        int kk = f4 >> 5;                        // 32 float4 per kk row
        int m  = (f4 & 31) * 4;
        const float4 v = *(const float4*)&A[(long)(kt + kk) * lda + m0 + m];
        sA[buf][(m + 0) * LS + kk] = f2bf(v.x);
        sA[buf][(m + 1) * LS + kk] = f2bf(v.y);
        sA[buf][(m + 2) * LS + kk] = f2bf(v.z);
        sA[buf][(m + 3) * LS + kk] = f2bf(v.w);
      }
    }
  };
  auto loadB = [&](int buf, int kt) {
    if (!transW) {                               // W(kk,n) = Wb[kk*ldw + n]
      #pragma unroll
      for (int c = 0; c < 4; ++c) {
        int f4 = tid + c * 256;
        int kk = f4 >> 5;
        int n  = (f4 & 31) * 4;
        const float4 v = *(const float4*)&W[(long)(kt + kk) * ldw + n0 + n];
        sB[buf][(n + 0) * LS + kk] = f2bf(v.x);
        sB[buf][(n + 1) * LS + kk] = f2bf(v.y);
        sB[buf][(n + 2) * LS + kk] = f2bf(v.z);
        sB[buf][(n + 3) * LS + kk] = f2bf(v.w);
      }
    } else {                                     // W(kk,n) = Wb[n*ldw + kk]
      #pragma unroll
      for (int c = 0; c < 4; ++c) {
        int f4 = tid + c * 256;
        int n  = f4 >> 3;
        int kk = (f4 & 7) * 4;
        const float4 v = *(const float4*)&W[(long)(n0 + n) * ldw + kt + kk];
        *(uint2*)&sB[buf][n * LS + kk] = make_uint2(pack2(v.x, v.y), pack2(v.z, v.w));
      }
    }
  };

  loadA(0, 0); loadB(0, 0);
  if (aMode == 2) asm volatile("s_wait_asynccnt 0x0" ::: "memory");
  __syncthreads();

  const int nk = Kdim / TK;
  for (int kb = 0; kb < nk; ++kb) {
    const int cur = kb & 1;
    if (kb + 1 < nk) {
      loadA(cur ^ 1, (kb + 1) * TK);
      loadB(cur ^ 1, (kb + 1) * TK);
      if (!transW && kb + 2 < nk)                 // hint HBM for the k-block after next
        __builtin_prefetch(&W[(long)((kb + 2) * TK + (tid >> 3)) * ldw + n0 + (tid & 7) * 16], 0, 1);
    }

    // Fragment gathers per documented CDNA5 WMMA VGPR layouts (ds_load_b128):
    // A 16x32 bf16: lanes 0-15 -> M=lane, K {0..7,16..23}; lanes 16-31 -> K {8..15,24..31}
    // B 32x16 bf16: lanes 0-15 -> N=lane, K {0..7,16..23}; lanes 16-31 -> N=lane-16, K {8..15,24..31}
    Frag32B af[4], bfg[2];
    const int rsel   = lane & 15;
    const int kaBase = (lane >= 16) ? 8 : 0;
    const int kbBase = (lane >= 16) ? 8 : 0;
    #pragma unroll
    for (int mt = 0; mt < 4; ++mt) {
      const unsigned short* p = &sA[cur][(wm + mt * 16 + rsel) * LS];
      af[mt].q[0] = *(const uint4*)(p + kaBase);
      af[mt].q[1] = *(const uint4*)(p + 16 + kaBase);
    }
    #pragma unroll
    for (int nt = 0; nt < 2; ++nt) {
      const unsigned short* p = &sB[cur][(wn + nt * 16 + rsel) * LS];
      bfg[nt].q[0] = *(const uint4*)(p + kbBase);
      bfg[nt].q[1] = *(const uint4*)(p + 16 + kbBase);
    }
    #pragma unroll
    for (int mt = 0; mt < 4; ++mt)
      #pragma unroll
      for (int nt = 0; nt < 2; ++nt)
        acc[mt][nt] = __builtin_amdgcn_wmma_f32_16x16x32_bf16(
            false, af[mt].v, false, bfg[nt].v, (short)0, acc[mt][nt], false, false);

    if (aMode == 2) asm volatile("s_wait_asynccnt 0x0" ::: "memory");
    __syncthreads();
  }

  // Epilogue: C layout lane(0-15)->N=lane,M=r ; lane(16-31)->N=lane-16,M=8+r
  float* Cf = (float*)Cb + (long)blockIdx.z * strideC;
  unsigned short* Ch = (unsigned short*)Cb + (long)blockIdx.z * strideC;
  #pragma unroll
  for (int mt = 0; mt < 4; ++mt) {
    #pragma unroll
    for (int nt = 0; nt < 2; ++nt) {
      const int col = n0 + wn + nt * 16 + (lane & 15);
      const float bv = bias ? bias[col] : 0.f;
      #pragma unroll
      for (int r = 0; r < 8; ++r) {
        const int row = m0 + wm + mt * 16 + ((lane < 16) ? r : r + 8);
        float v = acc[mt][nt][r] + bv;
        if (relu) v = fmaxf(v, 0.f);
        if (outBF16) {
          Ch[(long)row * ldc + col] = f2bf(v);
        } else {
          float* cp = &Cf[(long)row * ldc + col];
          if (accum) v += *cp;
          *cp = v;
        }
      }
    }
  }
}

// ---------------------------------------------------------------------------
// Small kernels
// ---------------------------------------------------------------------------
__global__ void gate_softmax_kernel(const float* __restrict__ g1,
                                    const float* __restrict__ Wg2,
                                    const float* __restrict__ bg2,
                                    float* __restrict__ gate) {
  int b = blockIdx.x * blockDim.x + threadIdx.x;
  if (b >= B_) return;
  float lg[K_];
  #pragma unroll
  for (int k = 0; k < K_; ++k) lg[k] = bg2[k];
  const float* row = g1 + (long)b * GH_;
  for (int j = 0; j < GH_; ++j) {
    float xv = row[j];
    const float* wr = Wg2 + (long)j * K_;
    #pragma unroll
    for (int k = 0; k < K_; ++k) lg[k] += xv * wr[k];
  }
  float mx = lg[0];
  #pragma unroll
  for (int k = 1; k < K_; ++k) mx = fmaxf(mx, lg[k]);
  float s = 0.f;
  #pragma unroll
  for (int k = 0; k < K_; ++k) { lg[k] = __expf(lg[k] - mx); s += lg[k]; }
  float inv = 1.f / s;
  #pragma unroll
  for (int k = 0; k < K_; ++k) gate[b * K_ + k] = lg[k] * inv;
}

__global__ void skew_kernel(const float* __restrict__ A, float* __restrict__ S) {
  int idx = blockIdx.x * blockDim.x + threadIdx.x;
  if (idx >= DO_ * DO_) return;
  int i = idx / DO_, j = idx % DO_;
  S[idx] = A[idx] - A[j * DO_ + i];
}

// Bt = transpose(I + 2S + S^2); init CG state (system (I - S^2) X = I+2S+S^2)
__global__ void cayley_rhs_kernel(const float* __restrict__ S, const float* __restrict__ S2,
                                  float* __restrict__ Bt, float* __restrict__ Xt,
                                  float* __restrict__ Rt, float* __restrict__ Pt) {
  int idx = blockIdx.x * blockDim.x + threadIdx.x;
  if (idx >= DO_ * DO_) return;
  int c = idx / DO_, d = idx % DO_;     // Bt[c][d] = RHS[d][c]
  float v = ((c == d) ? 1.f : 0.f) + 2.f * S[d * DO_ + c] + S2[c * DO_ + d]; // S2 symmetric
  Bt[idx] = v; Xt[idx] = 0.f; Rt[idx] = v; Pt[idx] = v;
}

__global__ void rowdot_kernel(const float* __restrict__ X, const float* __restrict__ Y,
                              float* __restrict__ out, int ncols) {
  __shared__ float red[256];
  int row = blockIdx.x;
  const float* xr = X + (long)row * ncols;
  const float* yr = Y + (long)row * ncols;
  float s = 0.f;
  for (int d = threadIdx.x; d < ncols; d += 256) s += xr[d] * yr[d];
  red[threadIdx.x] = s; __syncthreads();
  for (int o = 128; o > 0; o >>= 1) { if (threadIdx.x < o) red[threadIdx.x] += red[threadIdx.x + o]; __syncthreads(); }
  if (threadIdx.x == 0) out[row] = red[0];
}

__global__ void residualize_kernel(const float* __restrict__ Pt, float* __restrict__ Wt, int n) {
  int idx = blockIdx.x * blockDim.x + threadIdx.x;
  if (idx < n) Wt[idx] = Pt[idx] - Wt[idx];      // W = P*(I - S^2)  (given Wt = P*S*S)
}

__global__ void cg_updateA_kernel(const float* __restrict__ rho, const float* __restrict__ pw,
                                  const float* __restrict__ Pt, const float* __restrict__ Wt,
                                  float* __restrict__ Xt, float* __restrict__ Rt) {
  int row = blockIdx.x;
  float num = rho[row], den = pw[row];
  float alpha = (fabsf(den) > 1e-30f) ? num / den : 0.f;
  long base = (long)row * DO_;
  for (int c = threadIdx.x; c < DO_; c += 256) {
    Xt[base + c] += alpha * Pt[base + c];
    Rt[base + c] -= alpha * Wt[base + c];
  }
}

__global__ void cg_updateB_kernel(float* __restrict__ rho, const float* __restrict__ rho2,
                                  const float* __restrict__ Rt, float* __restrict__ Pt) {
  int row = blockIdx.x;
  float r0 = rho[row], r1 = rho2[row];
  float beta = (r0 > 1e-30f) ? r1 / r0 : 0.f;
  long base = (long)row * DO_;
  for (int c = threadIdx.x; c < DO_; c += 256) Pt[base + c] = Rt[base + c] + beta * Pt[base + c];
  __syncthreads();
  if (threadIdx.x == 0) rho[row] = r1;
}

// In-place: V -> V/clip(||V||,1e-6) ; NT -> second normalization (unit rows)
__global__ void rownorm_kernel(float* __restrict__ V, float* __restrict__ NT) {
  __shared__ float red[256];
  long base = (long)blockIdx.x * DO_;
  float s = 0.f;
  for (int d = threadIdx.x; d < DO_; d += 256) { float v = V[base + d]; s += v * v; }
  red[threadIdx.x] = s; __syncthreads();
  for (int o = 128; o > 0; o >>= 1) { if (threadIdx.x < o) red[threadIdx.x] += red[threadIdx.x + o]; __syncthreads(); }
  float n  = sqrtf(red[0]);
  float n1 = fmaxf(n, 1e-6f);
  float n2 = fmaxf(n / n1, 1e-12f);
  float i1 = 1.f / n1, i2 = i1 / n2;
  __syncthreads();
  for (int d = threadIdx.x; d < DO_; d += 256) {
    float v = V[base + d];
    V[base + d]  = v * i1;
    NT[base + d] = v * i2;
  }
}

__global__ void cosloss_kernel(const float* __restrict__ NT, float* __restrict__ accum) {
  __shared__ float red[256];
  int b = blockIdx.x;
  float local = 0.f;
  for (int k = 0; k < K_; ++k) {
    for (int l = k + 1; l < K_; ++l) {
      const float* a = NT + ((long)b * K_ + k) * DO_;
      const float* c = NT + ((long)b * K_ + l) * DO_;
      float s = 0.f;
      for (int d = threadIdx.x; d < DO_; d += 256) s += a[d] * c[d];
      red[threadIdx.x] = s; __syncthreads();
      for (int o = 128; o > 0; o >>= 1) { if (threadIdx.x < o) red[threadIdx.x] += red[threadIdx.x + o]; __syncthreads(); }
      if (threadIdx.x == 0) local += fabsf(red[0]);
      __syncthreads();
    }
  }
  if (threadIdx.x == 0) atomicAdd(accum, local * (2.f / (float)(K_ * (K_ - 1))) / (float)B_);
}

// Modified Gram-Schmidt QR per expert; columns of xt_k = nt[b,k,:] (contiguous).
__global__ __launch_bounds__(512) void qr_mgs_kernel(const float* __restrict__ NT,
                                                     float* __restrict__ Qall,
                                                     float* __restrict__ rdiag) {
  __shared__ float red[512];
  __shared__ float sval;
  const int k = blockIdx.x;
  float* Q = Qall + (long)k * B_ * DO_;
  const int tid = threadIdx.x, lane = tid & 31, wave = tid >> 5;

  for (int idx = tid; idx < B_ * DO_; idx += 512) {
    int b = idx / DO_, d = idx % DO_;
    Q[idx] = NT[((long)b * K_ + k) * DO_ + d];
  }
  __syncthreads();

  for (int j = 0; j < B_; ++j) {
    float* qj = Q + (long)j * DO_;
    float s = 0.f;
    for (int d = tid; d < DO_; d += 512) s += qj[d] * qj[d];
    red[tid] = s; __syncthreads();
    for (int o = 256; o > 0; o >>= 1) { if (tid < o) red[tid] += red[tid + o]; __syncthreads(); }
    if (tid == 0) {
      float r = sqrtf(red[0]);
      rdiag[k * B_ + j] = r;
      sval = (r > 1e-20f) ? 1.f / r : 0.f;
    }
    __syncthreads();
    float inv = sval;
    for (int d = tid; d < DO_; d += 512) qj[d] *= inv;
    __syncthreads();
    for (int i = j + 1 + wave; i < B_; i += 16) {
      float* qi = Q + (long)i * DO_;
      float dt = 0.f;
      for (int d = lane; d < DO_; d += 32) dt += qj[d] * qi[d];
      for (int o = 16; o > 0; o >>= 1) dt += __shfl_down(dt, o, 32);
      dt = __shfl(dt, 0, 32);
      for (int d = lane; d < DO_; d += 32) qi[d] -= dt * qj[d];
    }
    __syncthreads();
  }

  // numerical-rank mask: zero columns with index >= kcnt
  float cnt = 0.f;
  for (int j = tid; j < B_; j += 512) cnt += (fabsf(rdiag[k * B_ + j]) > 1e-3f) ? 1.f : 0.f;
  red[tid] = cnt; __syncthreads();
  for (int o = 256; o > 0; o >>= 1) { if (tid < o) red[tid] += red[tid + o]; __syncthreads(); }
  int kcnt = (int)(red[0] + 0.5f);
  for (int idx = tid; idx < B_ * DO_; idx += 512) {
    if ((idx / DO_) >= kcnt) Q[idx] = 0.f;
  }
}

__global__ void misc_init_kernel(float* __restrict__ y, float* __restrict__ accum) {
  int d = blockIdx.x * blockDim.x + threadIdx.x;
  if (d < DO_) y[d] = 0.03125f;            // 1/sqrt(1024)
  if (d == 0) *accum = 0.f;
}

__global__ void scale_kernel(float* __restrict__ X, float s, int n) {
  int idx = blockIdx.x * blockDim.x + threadIdx.x;
  if (idx < n) X[idx] *= s;
}

__global__ void gemv_kernel(const float* __restrict__ Mx, const float* __restrict__ y,
                            float* __restrict__ z) {
  __shared__ float red[256];
  int row = blockIdx.x;
  const float* mr = Mx + (long)row * DO_;
  float s = 0.f;
  for (int d = threadIdx.x; d < DO_; d += 256) s += mr[d] * y[d];
  red[threadIdx.x] = s; __syncthreads();
  for (int o = 128; o > 0; o >>= 1) { if (threadIdx.x < o) red[threadIdx.x] += red[threadIdx.x + o]; __syncthreads(); }
  if (threadIdx.x == 0) z[row] = red[0];
}

__global__ void normscale_kernel(const float* __restrict__ z, float* __restrict__ y) {
  __shared__ float red[256];
  float s = 0.f;
  for (int d = threadIdx.x; d < DO_; d += 256) s += z[d] * z[d];
  red[threadIdx.x] = s; __syncthreads();
  for (int o = 128; o > 0; o >>= 1) { if (threadIdx.x < o) red[threadIdx.x] += red[threadIdx.x + o]; __syncthreads(); }
  float inv = rsqrtf(fmaxf(red[0], 1e-30f));
  __syncthreads();
  for (int d = threadIdx.x; d < DO_; d += 256) y[d] = z[d] * inv;
}

__global__ void rayleigh_kernel(const float* __restrict__ y, const float* __restrict__ z,
                                float* __restrict__ outp) {
  __shared__ float red[256];
  float s = 0.f;
  for (int d = threadIdx.x; d < DO_; d += 256) s += y[d] * z[d];
  red[threadIdx.x] = s; __syncthreads();
  for (int o = 128; o > 0; o >>= 1) { if (threadIdx.x < o) red[threadIdx.x] += red[threadIdx.x + o]; __syncthreads(); }
  if (threadIdx.x == 0) *outp = red[0];
}

__global__ void write_scalar_kernel(const float* __restrict__ src, float* __restrict__ dst) {
  if (threadIdx.x == 0 && blockIdx.x == 0) *dst = *src;
}

__global__ void combine_head_kernel(const float* __restrict__ Vn, const float* __restrict__ gate,
                                    const float* __restrict__ Wo, const float* __restrict__ bo,
                                    float* __restrict__ outp) {
  __shared__ float red[256];
  int b = blockIdx.x;
  float g[K_];
  #pragma unroll
  for (int k = 0; k < K_; ++k) g[k] = gate[b * K_ + k];
  float s = 0.f;
  for (int d = threadIdx.x; d < DO_; d += 256) {
    float c = 0.f;
    #pragma unroll
    for (int k = 0; k < K_; ++k) c += g[k] * Vn[((long)b * K_ + k) * DO_ + d];
    s += c * Wo[d];
  }
  red[threadIdx.x] = s; __syncthreads();
  for (int o = 128; o > 0; o >>= 1) { if (threadIdx.x < o) red[threadIdx.x] += red[threadIdx.x + o]; __syncthreads(); }
  if (threadIdx.x == 0) outp[b] = red[0] + bo[0];
}

// ---------------------------------------------------------------------------
// Host launcher
// ---------------------------------------------------------------------------
static void gemm(hipStream_t st,
                 const void* A, long sA, int lda, int aMode,
                 const float* W, long sW, int ldw, int tW,
                 const float* bias, long sBias,
                 void* C, long sC, int ldc,
                 int M, int N, int Kd, int relu, int acc, int outBF16) {
  dim3 g(N / TN, M / TM, 1);
  wmma_gemm_kernel<<<g, dim3(256), 0, st>>>(A, sA, lda, aMode, W, sW, ldw, tW,
                                            bias, sBias, C, sC, ldc, M, N, Kd, relu, acc, outBF16);
}
static void gemmZ(hipStream_t st,
                  const void* A, long sA, int lda, int aMode,
                  const float* W, long sW, int ldw, int tW,
                  const float* bias, long sBias,
                  void* C, long sC, int ldc,
                  int M, int N, int Kd, int relu, int acc, int outBF16, int Z) {
  dim3 g(N / TN, M / TM, Z);
  wmma_gemm_kernel<<<g, dim3(256), 0, st>>>(A, sA, lda, aMode, W, sW, ldw, tW,
                                            bias, sBias, C, sC, ldc, M, N, Kd, relu, acc, outBF16);
}

extern "C" void kernel_launch(void* const* d_in, const int* in_sizes, int n_in,
                              void* d_out, int out_size, void* d_ws, size_t ws_size,
                              hipStream_t stream) {
  const float* x   = (const float*)d_in[0];
  const float* We1 = (const float*)d_in[1];
  const float* be1 = (const float*)d_in[2];
  const float* We2 = (const float*)d_in[3];
  const float* be2 = (const float*)d_in[4];
  const float* We3 = (const float*)d_in[5];
  const float* be3 = (const float*)d_in[6];
  const float* Wg1 = (const float*)d_in[7];
  const float* bg1 = (const float*)d_in[8];
  const float* Wg2 = (const float*)d_in[9];
  const float* bg2 = (const float*)d_in[10];
  const float* Amat= (const float*)d_in[11];
  const float* Wo  = (const float*)d_in[12];
  const float* bo  = (const float*)d_in[13];
  float* out = (float*)d_out;

  // carve workspace (256B aligned chunks)
  char* p = (char*)d_ws;
  auto allocB = [&](size_t bytes) { void* r = (void*)p; p += (bytes + 255) & ~(size_t)255; return r; };
  auto allocF = [&](size_t n) { return (float*)allocB(n * sizeof(float)); };
  auto allocH = [&](size_t n) { return (unsigned short*)allocB(n * sizeof(unsigned short)); };

  unsigned short* h1  = allocH((size_t)K_ * B_ * H1_);   // bf16 activations
  unsigned short* h2  = allocH((size_t)K_ * B_ * H2_);
  unsigned short* eo  = allocH((size_t)K_ * B_ * DO_);
  unsigned short* crd = allocH((size_t)B_ * K_ * SZ_);
  float* g1     = allocF((size_t)B_ * GH_);
  float* gate   = allocF((size_t)B_ * K_);
  float* S      = allocF((size_t)DO_ * DO_);
  float* S2     = allocF((size_t)DO_ * DO_);
  float* Bt     = allocF((size_t)DO_ * DO_);
  float* Xt     = allocF((size_t)DO_ * DO_);
  float* Rt     = allocF((size_t)DO_ * DO_);
  float* Pt     = allocF((size_t)DO_ * DO_);
  float* Tt     = allocF((size_t)DO_ * DO_);
  float* Wt     = allocF((size_t)DO_ * DO_);
  float* rho    = allocF(DO_);
  float* pw     = allocF(DO_);
  float* rho2   = allocF(DO_);
  float* V      = allocF((size_t)B_ * K_ * DO_);
  float* NT     = allocF((size_t)B_ * K_ * DO_);
  float* Qm     = allocF((size_t)K_ * B_ * DO_);
  float* rdiag  = allocF((size_t)K_ * B_);
  float* Pm     = allocF((size_t)DO_ * DO_);
  float* yv     = allocF(DO_);
  float* zv     = allocF(DO_);
  float* accum  = allocF(16);

  // ---- expert MLP (dominant FLOPs): fp32 weights converted in regs,
  //      bf16 activations DMA'd global->LDS (async), bf16 outputs ----
  gemmZ(stream, x,  0, D_, 0,  We1, (long)D_ * H1_, H1_, 0, be1, H1_,
        h1, (long)B_ * H1_, H1_, B_, H1_, D_, 1, 0, 1, K_);
  gemmZ(stream, h1, (long)B_ * H1_, H1_, 2, We2, (long)H1_ * H2_, H2_, 0, be2, H2_,
        h2, (long)B_ * H2_, H2_, B_, H2_, H1_, 1, 0, 1, K_);
  gemmZ(stream, h2, (long)B_ * H2_, H2_, 2, We3, (long)H2_ * DO_, DO_, 0, be3, DO_,
        eo, (long)B_ * DO_, DO_, B_, DO_, H2_, 0, 0, 1, K_);

  // ---- gating ----
  gemm(stream, x, 0, D_, 0, Wg1, 0, GH_, 0, bg1, 0, g1, 0, GH_, B_, GH_, D_, 1, 0, 0);
  gate_softmax_kernel<<<dim3(4), dim3(128), 0, stream>>>(g1, Wg2, bg2, gate);

  // ---- Cayley: (I - S^2) X = (I + S)^2 via block CG (SPD, cond <= 1+lmax(S'S)) ----
  skew_kernel<<<dim3((DO_ * DO_) / 256), dim3(256), 0, stream>>>(Amat, S);
  gemm(stream, S, 0, DO_, 0, S, 0, DO_, 0, nullptr, 0, S2, 0, DO_, DO_, DO_, DO_, 0, 0, 0);
  cayley_rhs_kernel<<<dim3((DO_ * DO_) / 256), dim3(256), 0, stream>>>(S, S2, Bt, Xt, Rt, Pt);
  rowdot_kernel<<<dim3(DO_), dim3(256), 0, stream>>>(Rt, Rt, rho, DO_);
  for (int it = 0; it < CG_ITERS; ++it) {
    gemm(stream, Pt, 0, DO_, 0, S, 0, DO_, 0, nullptr, 0, Tt, 0, DO_, DO_, DO_, DO_, 0, 0, 0);
    gemm(stream, Tt, 0, DO_, 0, S, 0, DO_, 0, nullptr, 0, Wt, 0, DO_, DO_, DO_, DO_, 0, 0, 0);
    residualize_kernel<<<dim3((DO_ * DO_) / 256), dim3(256), 0, stream>>>(Pt, Wt, DO_ * DO_);
    rowdot_kernel<<<dim3(DO_), dim3(256), 0, stream>>>(Pt, Wt, pw, DO_);
    cg_updateA_kernel<<<dim3(DO_), dim3(256), 0, stream>>>(rho, pw, Pt, Wt, Xt, Rt);
    rowdot_kernel<<<dim3(DO_), dim3(256), 0, stream>>>(Rt, Rt, rho2, DO_);
    cg_updateB_kernel<<<dim3(DO_), dim3(256), 0, stream>>>(rho, rho2, Rt, Pt);
  }
  // Xt now holds Qc transposed: Xt[c*DO + d] = Qc[d][c]

  // ---- subspace coords & projection (WMMA; A-side async bf16) ----
  // coords[b,k,:] = eo[k,b,:] @ Qc[:, kSZ:(k+1)SZ]  -> W(kk,n)=Xt[(kSZ+n)*DO+kk] (transW)
  gemmZ(stream, eo, (long)B_ * DO_, DO_, 2, Xt, (long)SZ_ * DO_, DO_, 1, nullptr, 0,
        crd, SZ_, K_ * SZ_, B_, SZ_, DO_, 0, 0, 1, K_);
  // V[b,k,:] = coords[b,k,:] @ Qblock^T -> W(kk,n)=Xt[(kSZ+kk)*DO+n] (no trans)
  gemmZ(stream, crd, SZ_, K_ * SZ_, 2, Xt, (long)SZ_ * DO_, DO_, 0, nullptr, 0,
        V, DO_, K_ * DO_, B_, DO_, SZ_, 0, 0, 0, K_);

  rownorm_kernel<<<dim3(B_ * K_), dim3(256), 0, stream>>>(V, NT);

  // ---- losses ----
  misc_init_kernel<<<dim3(4), dim3(256), 0, stream>>>(yv, accum);
  cosloss_kernel<<<dim3(B_), dim3(256), 0, stream>>>(NT, accum);

  qr_mgs_kernel<<<dim3(K_), dim3(512), 0, stream>>>(NT, Qm, rdiag);

  // Pm = mean_k Qm_k Qm_k^T : A = Qm_k (transA, col-major), W = Qm_k^T (row-major view)
  for (int k = 0; k < K_; ++k) {
    const float* Qk = Qm + (size_t)k * B_ * DO_;
    gemm(stream, Qk, 0, DO_, 1, Qk, 0, DO_, 0, nullptr, 0,
         Pm, 0, DO_, DO_, DO_, B_, 0, (k > 0) ? 1 : 0, 0);
  }
  scale_kernel<<<dim3((DO_ * DO_) / 256), dim3(256), 0, stream>>>(Pm, 1.f / (float)K_, DO_ * DO_);

  for (int it = 0; it < POW_ITERS; ++it) {
    gemv_kernel<<<dim3(DO_), dim3(256), 0, stream>>>(Pm, yv, zv);
    normscale_kernel<<<dim3(1), dim3(256), 0, stream>>>(zv, yv);
  }
  gemv_kernel<<<dim3(DO_), dim3(256), 0, stream>>>(Pm, yv, zv);
  rayleigh_kernel<<<dim3(1), dim3(256), 0, stream>>>(yv, zv, out + B_ + 1);   // lambda_loss
  write_scalar_kernel<<<dim3(1), dim3(1), 0, stream>>>(accum, out + B_);      // cosine_loss

  // ---- gated combine + head ----
  combine_head_kernel<<<dim3(B_), dim3(256), 0, stream>>>(V, gate, Wo, bo, out);
}
